// MoELayer_46608985096886
// MI455X (gfx1250) — compile-verified, matrix operand order checked by reading, not run on hardware
//
#include <hip/hip_runtime.h>

typedef __attribute__((ext_vector_type(16))) __bf16 v16bf;
typedef __attribute__((ext_vector_type(8)))  float  v8f;

constexpr int NTOK = 4096;   // B*T
constexpr int DIM  = 1024;   // D
constexpr int NEXP = 8;      // E
constexpr int HID  = 2048;   // H
constexpr int BM = 128;      // gathered rows per block
constexpr int BN = 64;       // output cols per block
constexpr int BK = 32;       // k-slice per stage
constexpr int LDK = BK + 8;  // padded LDS stride (elems) -> conflict-free b128 ds loads

__device__ __forceinline__ v8f zero8() {
  v8f z;
#pragma unroll
  for (int i = 0; i < 8; ++i) z[i] = 0.f;
  return z;
}

// CDNA5 async copy: global -> LDS, 16 bytes per lane, tracked by ASYNCcnt.
// VDST operand is the LDS byte offset (low 32 bits of the generic shared pointer).
__device__ __forceinline__ void asyncLdsB128(const void* gaddr, void* lds) {
  unsigned ldsOff = (unsigned)(unsigned long long)(uintptr_t)lds;
  unsigned long long ga = (unsigned long long)(uintptr_t)gaddr;
  asm volatile("global_load_async_to_lds_b128 %0, %1, off"
               :: "v"(ldsOff), "v"(ga)
               : "memory");
}
__device__ __forceinline__ void waitAsync() {
  asm volatile("s_wait_asynccnt 0x0" ::: "memory");
}

// Load a 16x32 bf16 A/B fragment from an LDS tile laid out row-major [rows][LDK].
// Per ISA 7.12.2: lane L<16 -> row rowBase+L, K 0..7 & 16..23 ; lane L>=16 -> K 8..15 & 24..31.
__device__ __forceinline__ v16bf ldsFrag(const __bf16* tile, int rowBase, int lane) {
  int r  = rowBase + (lane & 15);
  int k0 = (lane & 16) ? 8 : 0;
  const __bf16* p = tile + r * LDK + k0;
  v16bf f;
  uint4* u = reinterpret_cast<uint4*>(&f);
  u[0] = *reinterpret_cast<const uint4*>(p);        // K k0 .. k0+7
  u[1] = *reinterpret_cast<const uint4*>(p + 16);   // K k0+16 .. k0+23
  return f;
}

__device__ __forceinline__ v8f wmma_bf16(v16bf a, v16bf b, v8f c) {
  return __builtin_amdgcn_wmma_f32_16x16x32_bf16(false, a, false, b, (short)0, c, false, false);
}

// ---------------- Stage 1: gate -> softmax -> top-2 -> lists; also emit bf16 copy of x --
__global__ __launch_bounds__(256) void moe_gate(
    const float* __restrict__ x, const float* __restrict__ gw,
    int* __restrict__ cnt, int* __restrict__ tok, int* __restrict__ slotl,
    float* __restrict__ wts, __bf16* __restrict__ xbf)
{
  const int n    = blockIdx.x * 8 + (threadIdx.x >> 5);   // one wave32 per token
  const int lane = threadIdx.x & 31;
  float acc[NEXP];
#pragma unroll
  for (int e = 0; e < NEXP; ++e) acc[e] = 0.f;
  const float* xr = x + (size_t)n * DIM;
  __bf16* xo = xbf + (size_t)n * DIM;
  for (int k = lane; k < DIM; k += 32) {
    float xv = xr[k];
    xo[k] = (__bf16)xv;                                   // bf16 copy for WMMA A-tiles
#pragma unroll
    for (int e = 0; e < NEXP; ++e) acc[e] += xv * gw[e * DIM + k];
  }
#pragma unroll
  for (int e = 0; e < NEXP; ++e) {
#pragma unroll
    for (int off = 16; off > 0; off >>= 1)
      acc[e] += __shfl_xor(acc[e], off, 32);
  }
  if (lane == 0) {
    float mx = acc[0];
#pragma unroll
    for (int e = 1; e < NEXP; ++e) mx = fmaxf(mx, acc[e]);
    float p[NEXP], s = 0.f;
#pragma unroll
    for (int e = 0; e < NEXP; ++e) { p[e] = __expf(acc[e] - mx); s += p[e]; }
    int i0 = 0; float v0 = p[0];
#pragma unroll
    for (int e = 1; e < NEXP; ++e) if (p[e] > v0) { v0 = p[e]; i0 = e; }
    int i1 = -1; float v1 = -1.f;
#pragma unroll
    for (int e = 0; e < NEXP; ++e) if (e != i0 && p[e] > v1) { v1 = p[e]; i1 = e; }
    v0 /= s; v1 /= s;                    // top-2 softmax probs
    float inv = 1.f / (v0 + v1 + 1e-8f); // renormalize
    float w0 = v0 * inv, w1 = v1 * inv;
    int p0 = atomicAdd(cnt + i0, 1);
    tok[i0 * NTOK + p0]   = n;
    slotl[i0 * NTOK + p0] = 2 * n;
    wts[2 * n] = w0;
    int p1 = atomicAdd(cnt + i1, 1);
    tok[i1 * NTOK + p1]   = n;
    slotl[i1 * NTOK + p1] = 2 * n + 1;
    wts[2 * n + 1] = w1;
  }
}

// ---------------- Stage 2: gathered GEMM  a = silu(X W1^T) * (X W3^T)  (bf16 out) ------
// 128x64 block tile; 8 waves in 4(row)x2(col); each wave: 32x32 per matrix = 8 WMMA/step.
// A tile: async DMA (bf16 x). B tiles: fp32 stream + cvt (co-executes with WMMA).
// Double-buffered LDS, one barrier per k-step.
__global__ __launch_bounds__(256) void moe_ffn1(
    const __bf16* __restrict__ xbf,
    const float* __restrict__ w1, const float* __restrict__ w3,
    const int* __restrict__ cnt, const int* __restrict__ tok,
    const int* __restrict__ slotl, __bf16* __restrict__ act)
{
  const int e  = blockIdx.z;
  const int Me = cnt[e];
  const int m0 = blockIdx.y * BM;
  if (m0 >= Me) return;
  const int n0 = blockIdx.x * BN;            // within HID

  __shared__ __bf16 sA [2][BM][LDK];         // 20 KB
  __shared__ __bf16 sB1[2][BN][LDK];         // 10 KB
  __shared__ __bf16 sB3[2][BN][LDK];         // 10 KB

  const int tid  = threadIdx.x;
  const int lane = tid & 31;
  const int wave = tid >> 5;                 // 8 waves
  const int wm   = wave >> 1;                // 0..3 (32-row slice)
  const int wn   = wave & 1;                 // 0..1 (32-col slice)

  const int arow = tid >> 2;                 // 0..63
  const int aseg = (tid & 3) * 8;            // 0,8,16,24 (elems; *2 = bytes, 16B aligned)

  int aidx0 = m0 + arow;      if (aidx0 >= Me) aidx0 = Me - 1;  // clamp gather
  int aidx1 = m0 + arow + 64; if (aidx1 >= Me) aidx1 = Me - 1;
  const int gtok0 = tok[e * NTOK + aidx0];
  const int gtok1 = tok[e * NTOK + aidx1];
  const __bf16* agp0 = xbf + (size_t)gtok0 * DIM + aseg;
  const __bf16* agp1 = xbf + (size_t)gtok1 * DIM + aseg;
  const float*  b1p  = w1 + ((size_t)e * HID + n0 + arow) * DIM + aseg;
  const float*  b3p  = w3 + ((size_t)e * HID + n0 + arow) * DIM + aseg;

  v8f hacc[2][2] = {{zero8(), zero8()}, {zero8(), zero8()}};
  v8f gacc[2][2] = {{zero8(), zero8()}, {zero8(), zero8()}};

  auto loadStage = [&](int k0, int s) {
    asyncLdsB128(agp0 + k0, &sA[s][arow][aseg]);           // ASYNCcnt-tracked DMA
    asyncLdsB128(agp1 + k0, &sA[s][arow + 64][aseg]);
    float4 c0 = *(const float4*)(b1p + k0);
    float4 c1 = *(const float4*)(b1p + k0 + 4);
    float4 d0 = *(const float4*)(b3p + k0);
    float4 d1 = *(const float4*)(b3p + k0 + 4);
    if (k0 + BK < DIM) __builtin_prefetch(b1p + k0 + BK, 0, 1);
    __bf16 cb[8] = {(__bf16)c0.x,(__bf16)c0.y,(__bf16)c0.z,(__bf16)c0.w,
                    (__bf16)c1.x,(__bf16)c1.y,(__bf16)c1.z,(__bf16)c1.w};
    __bf16 db[8] = {(__bf16)d0.x,(__bf16)d0.y,(__bf16)d0.z,(__bf16)d0.w,
                    (__bf16)d1.x,(__bf16)d1.y,(__bf16)d1.z,(__bf16)d1.w};
    *(uint4*)&sB1[s][arow][aseg] = *(uint4*)cb;
    *(uint4*)&sB3[s][arow][aseg] = *(uint4*)db;
  };

  loadStage(0, 0);
  waitAsync();
  __syncthreads();

  for (int k0 = 0; k0 < DIM; k0 += BK) {
    const int cur = (k0 >> 5) & 1;
    if (k0 + BK < DIM) loadStage(k0 + BK, cur ^ 1);        // prefetch next stage

    v16bf af[2], b1f[2], b3f[2];
#pragma unroll
    for (int i = 0; i < 2; ++i) {
      af [i] = ldsFrag(&sA [cur][0][0], wm * 32 + i * 16, lane);
      b1f[i] = ldsFrag(&sB1[cur][0][0], wn * 32 + i * 16, lane);
      b3f[i] = ldsFrag(&sB3[cur][0][0], wn * 32 + i * 16, lane);
    }
#pragma unroll
    for (int mi = 0; mi < 2; ++mi) {
#pragma unroll
      for (int ni = 0; ni < 2; ++ni) {
        hacc[mi][ni] = wmma_bf16(af[mi], b1f[ni], hacc[mi][ni]);
        gacc[mi][ni] = wmma_bf16(af[mi], b3f[ni], gacc[mi][ni]);
      }
    }
    waitAsync();                                           // next stage's DMA landed
    __syncthreads();                                       // + everyone's cvt stores
  }

  const int rsub = (lane >> 4) << 3;                       // 0 or 8
  const int csub = lane & 15;
#pragma unroll
  for (int mi = 0; mi < 2; ++mi) {
#pragma unroll
    for (int ni = 0; ni < 2; ++ni) {
#pragma unroll
      for (int r = 0; r < 8; ++r) {
        int gm = m0 + wm * 32 + mi * 16 + rsub + r;
        if (gm < Me) {
          int slot = slotl[e * NTOK + gm];
          int col  = n0 + wn * 32 + ni * 16 + csub;
          float h = hacc[mi][ni][r], g = gacc[mi][ni][r];
          float a = (h / (1.f + __expf(-h))) * g;          // silu(h) * g
          act[(size_t)slot * HID + col] = (__bf16)a;
        }
      }
    }
  }
}

// ---------------- Stage 3: gathered GEMM  y = a W2^T  (fp32 out, per slot) -------------
__global__ __launch_bounds__(256) void moe_ffn2(
    const __bf16* __restrict__ act, const float* __restrict__ w2,
    const int* __restrict__ cnt, const int* __restrict__ slotl,
    float* __restrict__ y)
{
  const int e  = blockIdx.z;
  const int Me = cnt[e];
  const int m0 = blockIdx.y * BM;
  if (m0 >= Me) return;
  const int n0 = blockIdx.x * BN;            // within DIM

  __shared__ __bf16 sA[2][BM][LDK];
  __shared__ __bf16 sB[2][BN][LDK];

  const int tid  = threadIdx.x;
  const int lane = tid & 31;
  const int wave = tid >> 5;
  const int wm   = wave >> 1;                // 0..3
  const int wn   = wave & 1;                 // 0..1
  const int arow = tid >> 2;
  const int aseg = (tid & 3) * 8;

  int aidx0 = m0 + arow;      if (aidx0 >= Me) aidx0 = Me - 1;
  int aidx1 = m0 + arow + 64; if (aidx1 >= Me) aidx1 = Me - 1;
  const int gslot0 = slotl[e * NTOK + aidx0];
  const int gslot1 = slotl[e * NTOK + aidx1];
  const __bf16* agp0 = act + (size_t)gslot0 * HID + aseg;
  const __bf16* agp1 = act + (size_t)gslot1 * HID + aseg;
  const float*  brp  = w2 + ((size_t)e * DIM + n0 + arow) * HID + aseg;

  v8f acc[2][2] = {{zero8(), zero8()}, {zero8(), zero8()}};

  auto loadStage = [&](int k0, int s) {
    asyncLdsB128(agp0 + k0, &sA[s][arow][aseg]);           // bf16 activations: pure DMA
    asyncLdsB128(agp1 + k0, &sA[s][arow + 64][aseg]);
    float4 c0 = *(const float4*)(brp + k0);
    float4 c1 = *(const float4*)(brp + k0 + 4);
    if (k0 + BK < HID) __builtin_prefetch(brp + k0 + BK, 0, 1);
    __bf16 cb[8] = {(__bf16)c0.x,(__bf16)c0.y,(__bf16)c0.z,(__bf16)c0.w,
                    (__bf16)c1.x,(__bf16)c1.y,(__bf16)c1.z,(__bf16)c1.w};
    *(uint4*)&sB[s][arow][aseg] = *(uint4*)cb;
  };

  loadStage(0, 0);
  waitAsync();
  __syncthreads();

  for (int k0 = 0; k0 < HID; k0 += BK) {
    const int cur = (k0 >> 5) & 1;
    if (k0 + BK < HID) loadStage(k0 + BK, cur ^ 1);

    v16bf af[2], bf[2];
#pragma unroll
    for (int i = 0; i < 2; ++i) {
      af[i] = ldsFrag(&sA[cur][0][0], wm * 32 + i * 16, lane);
      bf[i] = ldsFrag(&sB[cur][0][0], wn * 32 + i * 16, lane);
    }
#pragma unroll
    for (int mi = 0; mi < 2; ++mi)
#pragma unroll
      for (int ni = 0; ni < 2; ++ni)
        acc[mi][ni] = wmma_bf16(af[mi], bf[ni], acc[mi][ni]);

    waitAsync();
    __syncthreads();
  }

  const int rsub = (lane >> 4) << 3;
  const int csub = lane & 15;
#pragma unroll
  for (int mi = 0; mi < 2; ++mi) {
#pragma unroll
    for (int ni = 0; ni < 2; ++ni) {
#pragma unroll
      for (int r = 0; r < 8; ++r) {
        int gm = m0 + wm * 32 + mi * 16 + rsub + r;
        if (gm < Me) {
          int slot = slotl[e * NTOK + gm];
          int col  = n0 + wn * 32 + ni * 16 + csub;
          y[(size_t)slot * DIM + col] = acc[mi][ni][r];
        }
      }
    }
  }
}

// ---------------- Stage 4: out[n] = w0*y[2n] + w1*y[2n+1] ------------------------------
__global__ __launch_bounds__(256) void moe_combine(
    const float* __restrict__ y, const float* __restrict__ wts, float* __restrict__ out)
{
  constexpr int D4 = DIM / 4;
  int i = blockIdx.x * 256 + threadIdx.x;        // over NTOK*DIM/4
  int n = i / D4, c = i - n * D4;
  const float4* y4 = reinterpret_cast<const float4*>(y);
  float4 a = y4[(size_t)(2 * n) * D4 + c];
  float4 b = y4[(size_t)(2 * n + 1) * D4 + c];
  float w0 = wts[2 * n], w1 = wts[2 * n + 1];
  float4 o;
  o.x = w0 * a.x + w1 * b.x;
  o.y = w0 * a.y + w1 * b.y;
  o.z = w0 * a.z + w1 * b.z;
  o.w = w0 * a.w + w1 * b.w;
  reinterpret_cast<float4*>(out)[i] = o;
}

extern "C" void kernel_launch(void* const* d_in, const int* in_sizes, int n_in,
                              void* d_out, int out_size, void* d_ws, size_t ws_size,
                              hipStream_t stream) {
  const float* x  = (const float*)d_in[0];
  const float* gw = (const float*)d_in[1];
  const float* w1 = (const float*)d_in[2];
  const float* w2 = (const float*)d_in[3];
  const float* w3 = (const float*)d_in[4];
  float* out = (float*)d_out;

  char* ws = (char*)d_ws;
  size_t off = 0;
  auto alloc = [&](size_t bytes) -> void* {
    off = (off + 255) & ~size_t(255);
    void* p = ws + off;
    off += bytes;
    return p;
  };
  int*    cnt   = (int*)   alloc(NEXP * sizeof(int));
  int*    tok   = (int*)   alloc((size_t)NEXP * NTOK * sizeof(int));
  int*    slotl = (int*)   alloc((size_t)NEXP * NTOK * sizeof(int));
  float*  wts   = (float*) alloc((size_t)2 * NTOK * sizeof(float));
  __bf16* xbf   = (__bf16*)alloc((size_t)NTOK * DIM * sizeof(__bf16));      //  8 MB
  __bf16* act   = (__bf16*)alloc((size_t)2 * NTOK * HID * sizeof(__bf16));  // 32 MB
  float*  ybuf  = (float*) alloc((size_t)2 * NTOK * DIM * sizeof(float));   // 32 MB

  hipMemsetAsync(cnt, 0, NEXP * sizeof(int), stream);
  moe_gate<<<NTOK / 8, 256, 0, stream>>>(x, gw, cnt, tok, slotl, wts, xbf);
  moe_ffn1<<<dim3(HID / BN, NTOK / BM, NEXP), 256, 0, stream>>>(xbf, w1, w3, cnt, tok, slotl, act);
  moe_ffn2<<<dim3(DIM / BN, NTOK / BM, NEXP), 256, 0, stream>>>(act, w2, cnt, slotl, ybuf);
  moe_combine<<<(NTOK * DIM / 4) / 256, 256, 0, stream>>>(ybuf, wts, out);
}
